// Multi_head_focus_attention_7980049236483
// MI455X (gfx1250) — matrix operand order repaired and checked by reading, not compiled, not run on hardware
//
#include <hip/hip_runtime.h>
#include <hip/hip_bf16.h>

typedef __attribute__((ext_vector_type(16))) _Float16 v16h;
typedef __attribute__((ext_vector_type(8)))  float    v8f;
typedef __attribute__((ext_vector_type(4)))  float    v4f;
typedef __attribute__((ext_vector_type(4)))  uint     v4u;

#define DM   256          // d_model
#define HW   48           // spatial height/width
#define SP   (HW * HW)    // 2304 positions per batch
#define NB   2            // batch
#define MTOT (NB * SP)    // 4608 GEMM rows
#define MT   32           // M rows per block (SP % 32 == 0 -> no batch straddle)
#define LDA  132          // LDS A-panel row stride in dwords:
                          //  132 % 64 == 4  -> conflict-free b128 fragment reads
                          //  132 * 4 = 528B -> every fragment chunk 16B aligned

union HPK { _Float16 h[2]; uint u; };
union H16 { v4u q[2]; v16h h; };

// A fragment (16x32 f16, MxK) per cdna5 ISA 7.12.2: lane m = lane&15, hi = lane>>4.
// Halves at K = [k0 + hi*8, +8) and [k0 + 16 + hi*8, +8) -> two 16B LDS b128 loads.
__device__ __forceinline__ v16h load_a_frag(const uint* Alds, int row16, int lane, int k0) {
    const int m  = (lane & 15) + row16;
    const int hi = lane >> 4;
    const uint* p = Alds + m * LDA + (k0 >> 1) + hi * 4;
    H16 cv;
    cv.q[0] = *(const v4u*)(p);
    cv.q[1] = *(const v4u*)(p + 8);
    return cv.h;
}

// WMMA core: 32-row A panel (LDS) x full N=256 weights. Wave w owns N tiles at
// n0 = w*16 and n0+128; 2 M sub-tiles x 2 N tiles = 4 accumulators, so each
// B fragment (two 16B global loads: lane col = lane&15, K offsets hi*8 and 16+hi*8)
// feeds two WMMAs.
__device__ __forceinline__ void wmma_core(const uint* Alds, const _Float16* __restrict__ Wh,
                                          int tid, v8f& c00, v8f& c01, v8f& c10, v8f& c11,
                                          int& n0a, int& n0b, int& hi, int& ncol) {
    const int wave = tid >> 5;
    const int lane = tid & 31;
    hi   = lane >> 4;
    ncol = lane & 15;
    n0a  = wave * 16;
    n0b  = n0a + 128;
    const _Float16* wa = Wh + (size_t)(n0a + ncol) * DM + hi * 8;
    const _Float16* wb = wa + (size_t)128 * DM;
    __builtin_prefetch(wa, 0, 1);
    __builtin_prefetch(wb, 0, 1);
#pragma unroll
    for (int kk = 0; kk < DM; kk += 32) {
        H16 hb0, hb1;
        hb0.q[0] = *(const v4u*)(wa + kk); hb0.q[1] = *(const v4u*)(wa + kk + 16);
        hb1.q[0] = *(const v4u*)(wb + kk); hb1.q[1] = *(const v4u*)(wb + kk + 16);
        v16h b0 = hb0.h;
        v16h b1 = hb1.h;
        v16h a0 = load_a_frag(Alds, 0,  lane, kk);
        v16h a1 = load_a_frag(Alds, 16, lane, kk);
        c00 = __builtin_amdgcn_wmma_f32_16x16x32_f16(false, a0, false, b0, (short)0, c00, false, false);
        c01 = __builtin_amdgcn_wmma_f32_16x16x32_f16(false, a0, false, b1, (short)0, c01, false, false);
        c10 = __builtin_amdgcn_wmma_f32_16x16x32_f16(false, a1, false, b0, (short)0, c10, false, false);
        c11 = __builtin_amdgcn_wmma_f32_16x16x32_f16(false, a1, false, b1, (short)0, c11, false, false);
    }
}

// Convert the four 256x256 f32 weight matrices to f16 once.
__global__ void __launch_bounds__(256)
cvt_weights_kernel(const float* __restrict__ Wq, const float* __restrict__ Wk,
                   const float* __restrict__ Wv, const float* __restrict__ Wfc,
                   _Float16* __restrict__ Wh) {
    const int idx = (blockIdx.x * 256 + threadIdx.x) * 4;   // 256 blocks cover 4*65536
    const int mi  = idx >> 16;
    const int off = idx & 65535;
    const float* src = (mi == 0) ? Wq : (mi == 1) ? Wk : (mi == 2) ? Wv : Wfc;
    v4f x = *(const v4f*)(src + off);
    _Float16* dst = Wh + (size_t)mi * 65536 + off;
    dst[0] = (_Float16)x[0]; dst[1] = (_Float16)x[1];
    dst[2] = (_Float16)x[2]; dst[3] = (_Float16)x[3];
}

// QKV projection GEMM: Out[m][o] = sum_c X[b][c][s] * W[o][c],   m=(b,s).
// X is NCHW f32 (channel stride SP); output row-major [m][256].
__global__ void __launch_bounds__(256)
proj_gemm_kernel(const float* __restrict__ X, const _Float16* __restrict__ Wh,
                 float* __restrict__ Out) {
    __shared__ uint Alds[MT * LDA];
    const int m0 = blockIdx.x * MT;
    const int b  = m0 / SP;
    const int s0 = m0 - b * SP;
    const int t  = threadIdx.x;

    {   // stage A panel [32 m x 256 k], f32 -> packed f16 pairs.
        // lanes 0..31 -> consecutive s (128B coalesced); one k-pair index per wave.
        const int r  = t & 31;
        const int kp = t >> 5;               // 0..7
        const float* src = X + (size_t)b * DM * SP + s0 + r;
#pragma unroll
        for (int j = 0; j < 16; ++j) {
            const int k2 = kp * 16 + j;      // dword index 0..127
            const int k  = k2 * 2;
            const float x0 = src[(size_t)k * SP];
            const float x1 = src[(size_t)(k + 1) * SP];
            HPK p; p.h[0] = (_Float16)x0; p.h[1] = (_Float16)x1;
            Alds[r * LDA + k2] = p.u;
        }
    }
    __syncthreads();

    v8f c00 = {}, c01 = {}, c10 = {}, c11 = {};
    int n0a, n0b, hi, ncol;
    wmma_core(Alds, Wh, t, c00, c01, c10, c11, n0a, n0b, hi, ncol);

    float* out = Out + (size_t)m0 * DM;      // D tile: row = v + 8*hi, col = ncol
#pragma unroll
    for (int v = 0; v < 8; ++v) {
        const int ml = v + hi * 8;
        out[(size_t)ml * DM + n0a + ncol]        = c00[v];
        out[(size_t)ml * DM + n0b + ncol]        = c01[v];
        out[(size_t)(ml + 16) * DM + n0a + ncol] = c10[v];
        out[(size_t)(ml + 16) * DM + n0b + ncol] = c11[v];
    }
}

// Local 7x7 windowed attention per (b, head, pos); zero-padded OOB neighbors
// contribute score 0 to the softmax and zero V, matching the reference exactly.
__global__ void __launch_bounds__(256)
focus_attn_kernel(const float* __restrict__ Qh, const float* __restrict__ Kp,
                  const float* __restrict__ Vp, float* __restrict__ Att) {
    const int g    = blockIdx.x * 256 + threadIdx.x;   // 0 .. 36863
    const int b    = g / (8 * SP);
    const int rem  = g - b * 8 * SP;
    const int head = rem / SP;
    const int s    = rem - head * SP;
    const int h    = s / HW;
    const int w    = s - h * HW;
    const size_t rowbase = (size_t)b * SP * DM + head * 32;

    const float* qr = Qh + rowbase + (size_t)s * DM;
    float qv[32];
#pragma unroll
    for (int d = 0; d < 32; ++d) qv[d] = qr[d];

    const float invt = 0.17677669529663689f;  // 1/sqrt(D_K=32)
    float sc[49];
#pragma unroll
    for (int ky = 0; ky < 7; ++ky) {
        const int hy = h + ky - 3;
#pragma unroll
        for (int kx = 0; kx < 7; ++kx) {
            const int wx = w + kx - 3;
            float dot = 0.f;
            if (hy >= 0 && hy < HW && wx >= 0 && wx < HW) {
                const float* kr = Kp + rowbase + (size_t)(hy * HW + wx) * DM;
#pragma unroll
                for (int d = 0; d < 32; ++d) dot += qv[d] * kr[d];
            }
            sc[ky * 7 + kx] = dot * invt;     // OOB -> 0, still in softmax
        }
    }

    float mx = sc[0];
#pragma unroll
    for (int j = 1; j < 49; ++j) mx = fmaxf(mx, sc[j]);
    float den = 0.f;
#pragma unroll
    for (int j = 0; j < 49; ++j) { sc[j] = __expf(sc[j] - mx); den += sc[j]; }
    const float inv = 1.f / den;

    float acc[32];
#pragma unroll
    for (int d = 0; d < 32; ++d) acc[d] = 0.f;
#pragma unroll
    for (int ky = 0; ky < 7; ++ky) {
        const int hy = h + ky - 3;
#pragma unroll
        for (int kx = 0; kx < 7; ++kx) {
            const int wx = w + kx - 3;
            if (hy >= 0 && hy < HW && wx >= 0 && wx < HW) {
                const float p = sc[ky * 7 + kx] * inv;
                const float* vr = Vp + rowbase + (size_t)(hy * HW + wx) * DM;
#pragma unroll
                for (int d = 0; d < 32; ++d) acc[d] += p * vr[d];
            }
        }
    }

    float* orow = Att + rowbase + (size_t)s * DM;
#pragma unroll
    for (int d = 0; d < 32; ++d) orow[d] = acc[d];
}

// Output fc GEMM + residual: out[b][o][s] = sum_k Att[m][k]*Wfc[o][k] + q[b][o][s].
__global__ void __launch_bounds__(256)
fc_gemm_kernel(const float* __restrict__ Att, const _Float16* __restrict__ Wh,
               const float* __restrict__ Qin, float* __restrict__ Out) {
    __shared__ uint Alds[MT * LDA];
    const int m0 = blockIdx.x * MT;
    const int t  = threadIdx.x;

    {   // A panel is row-major -> fully coalesced float4 staging (8192 floats)
        const float* src = Att + (size_t)m0 * DM;
#pragma unroll
        for (int it = 0; it < 8; ++it) {
            const int flat = (t + it * 256) * 4;     // 0 .. 8191
            v4f x = *(const v4f*)(src + flat);
            const int r  = flat >> 8;
            const int k2 = (flat & 255) >> 1;
            HPK p0, p1;
            p0.h[0] = (_Float16)x[0]; p0.h[1] = (_Float16)x[1];
            p1.h[0] = (_Float16)x[2]; p1.h[1] = (_Float16)x[3];
            Alds[r * LDA + k2]     = p0.u;
            Alds[r * LDA + k2 + 1] = p1.u;
        }
    }
    __syncthreads();

    v8f c00 = {}, c01 = {}, c10 = {}, c11 = {};
    int n0a, n0b, hi, ncol;
    wmma_core(Alds, Wh, t, c00, c01, c10, c11, n0a, n0b, hi, ncol);

    const int b  = m0 / SP;
    const int s0 = m0 - b * SP;
#pragma unroll
    for (int v = 0; v < 8; ++v) {
        const int s  = s0 + v + hi * 8;
        const size_t ia0 = ((size_t)b * DM + (n0a + ncol)) * SP + s;
        const size_t ib0 = ((size_t)b * DM + (n0b + ncol)) * SP + s;
        Out[ia0]      = c00[v] + Qin[ia0];
        Out[ib0]      = c01[v] + Qin[ib0];
        Out[ia0 + 16] = c10[v] + Qin[ia0 + 16];
        Out[ib0 + 16] = c11[v] + Qin[ib0 + 16];
    }
}

extern "C" void kernel_launch(void* const* d_in, const int* in_sizes, int n_in,
                              void* d_out, int out_size, void* d_ws, size_t ws_size,
                              hipStream_t stream) {
    const float* q   = (const float*)d_in[0];
    const float* k   = (const float*)d_in[1];
    const float* v   = (const float*)d_in[2];
    const float* Wq  = (const float*)d_in[3];
    const float* Wk  = (const float*)d_in[4];
    const float* Wv  = (const float*)d_in[5];
    const float* Wfc = (const float*)d_in[6];
    (void)in_sizes; (void)n_in; (void)out_size; (void)ws_size;

    // Workspace layout: [4 x 65536 f16 weights][qh][kp][vp][att] (each f32 [4608][256])
    _Float16* Wh = (_Float16*)d_ws;
    float* qh  = (float*)((char*)d_ws + 4 * 65536 * sizeof(_Float16));
    float* kp  = qh + (size_t)MTOT * DM;
    float* vp  = kp + (size_t)MTOT * DM;
    float* att = vp + (size_t)MTOT * DM;

    cvt_weights_kernel<<<256, 256, 0, stream>>>(Wq, Wk, Wv, Wfc, Wh);
    proj_gemm_kernel<<<MTOT / MT, 256, 0, stream>>>(q, Wh,             qh);
    proj_gemm_kernel<<<MTOT / MT, 256, 0, stream>>>(k, Wh + 65536,     kp);
    proj_gemm_kernel<<<MTOT / MT, 256, 0, stream>>>(v, Wh + 2 * 65536, vp);
    focus_attn_kernel<<<(NB * 8 * SP) / 256, 256, 0, stream>>>(qh, kp, vp, att);
    fc_gemm_kernel<<<MTOT / MT, 256, 0, stream>>>(att, Wh + 3 * 65536, q, (float*)d_out);
}
